// LlamaStandardAttn_32676111188731
// MI455X (gfx1250) — matrix operand order repaired and checked by reading, not compiled, not run on hardware
//
#include <hip/hip_runtime.h>
#include <hip/hip_bf16.h>

// ---------------------------------------------------------------------------
// Flash attention, B=2, S=2048, H=16, D=128, fp32 in/out, wave32 CDNA5.
// f16 WMMA (V_WMMA_F32_16X16X32_F16) with fp32 accumulation.
// 256-thread blocks (8 waves); each wave owns a 16-row Q tile; all waves
// share double-buffered 32-key K/V chunks in LDS. Softmax row reductions use
// DPP16 row_ror rotate-butterflies (pure VALU, no ds_bpermute).
// ---------------------------------------------------------------------------

typedef _Float16 v16h __attribute__((ext_vector_type(16)));
typedef _Float16 v8h  __attribute__((ext_vector_type(8)));
typedef _Float16 v4h  __attribute__((ext_vector_type(4)));
typedef float    v8f  __attribute__((ext_vector_type(8)));
typedef float    v4f  __attribute__((ext_vector_type(4)));

#define BB 2
#define SS 2048
#define HH 16
#define DD 128
#define KC 32              // keys per chunk
#define NCHUNK (SS / KC)   // 64
#define NWAVE 8            // waves per block
#define QROWS (NWAVE * 16) // 128 q rows per block

// LDS row strides (halves): padded for conflict-free ds_load_b128, rows stay
// 16B aligned (272B = 17*16, 80B = 5*16).
#define SK_STRIDE (DD + 8) // 136
#define SV_STRIDE 40
#define SP_STRIDE 40

__device__ __forceinline__ v16h cat8(v8h lo, v8h hi) {
  return __builtin_shufflevector(lo, hi, 0, 1, 2, 3, 4, 5, 6, 7,
                                 8, 9, 10, 11, 12, 13, 14, 15);
}

// DPP16 ROW_ROR:n rotate within each 16-lane row (wave32: two rows).
template <int N>
__device__ __forceinline__ float rot16(float x) {
  return __builtin_bit_cast(
      float, __builtin_amdgcn_update_dpp(0, __builtin_bit_cast(int, x),
                                         0x120 | N, 0xf, 0xf, true));
}
__device__ __forceinline__ float rowmax16(float x) {
  x = fmaxf(x, rot16<1>(x));
  x = fmaxf(x, rot16<2>(x));
  x = fmaxf(x, rot16<4>(x));
  x = fmaxf(x, rot16<8>(x));
  return x;
}
__device__ __forceinline__ float rowsum16(float x) {
  x += rot16<1>(x);
  x += rot16<2>(x);
  x += rot16<4>(x);
  x += rot16<8>(x);
  return x;
}

__global__ __launch_bounds__(256) void
fa_fwd_f16wmma(const float* __restrict__ q, const float* __restrict__ k,
               const float* __restrict__ v, const unsigned char* __restrict__ kpm,
               float* __restrict__ out) {
  __shared__ __align__(16) _Float16 sK[2][KC][SK_STRIDE];    // [buf][key][d]
  __shared__ __align__(16) _Float16 sVt[2][DD][SV_STRIDE];   // [buf][d][key]
  __shared__ __align__(16) _Float16 sP[NWAVE][16][SP_STRIDE];

  const int tid  = threadIdx.x;
  const int wave = tid >> 5;
  const int lane = tid & 31;
  const int half = lane >> 4;
  const int n16  = lane & 15;

  const int bh   = blockIdx.x >> 4;   // 16 q-blocks (of 128 rows) per (b,h)
  const int qblk = blockIdx.x & 15;
  const int b    = bh >> 4;           // H = 16
  const int h    = bh & 15;
  const int q0   = qblk * QROWS + wave * 16;

  const size_t rowstride = (size_t)HH * DD;
  const float* kbase = k + (size_t)b * SS * rowstride + (size_t)h * DD;
  const float* vbase = v + (size_t)b * SS * rowstride + (size_t)h * DD;

  // softmax scale 1/sqrt(128) with log2(e) folded in (exp -> exp2)
  const float qs = 0.08838834764831845f * 1.44269504088896340f;

  // ---- Q A-fragments (16-bit A 16x32 layout) ----
  v16h aq[4];
  {
    const float* qrow = q + ((size_t)(b * SS + q0 + n16) * HH + h) * DD;
#pragma unroll
    for (int c = 0; c < 4; ++c) {
      const float* p0 = qrow + c * 32 + 8 * half;
      v4f a0 = *(const v4f*)(p0);
      v4f a1 = *(const v4f*)(p0 + 4);
      v4f b0 = *(const v4f*)(p0 + 16);
      v4f b1 = *(const v4f*)(p0 + 20);
      v16h f;
#pragma unroll
      for (int e = 0; e < 4; ++e) {
        f[e]      = (_Float16)(a0[e] * qs);
        f[4 + e]  = (_Float16)(a1[e] * qs);
        f[8 + e]  = (_Float16)(b0[e] * qs);
        f[12 + e] = (_Float16)(b1[e] * qs);
      }
      aq[c] = f;
    }
  }

  v8f o[8];
#pragma unroll
  for (int t = 0; t < 8; ++t) o[t] = (v8f)0.0f;
  float rmax[8], rsum[8];
#pragma unroll
  for (int r = 0; r < 8; ++r) { rmax[r] = -__builtin_inff(); rsum[r] = 0.0f; }

  // ---- prologue: stage chunk 0 into buffer 0 ----
  // 32 keys * 128 d = 4096 f32 per tensor; 256 threads * 4 float4 each.
#pragma unroll
  for (int i = 0; i < 4; ++i) {
    int vec = tid + i * 256;
    int key = vec >> 5;
    int dv  = (vec & 31) << 2;
    v4f kv = *(const v4f*)(kbase + (size_t)key * rowstride + dv);
    v4f vv = *(const v4f*)(vbase + (size_t)key * rowstride + dv);
    v4h kh;
#pragma unroll
    for (int j = 0; j < 4; ++j) kh[j] = (_Float16)kv[j];
    *(v4h*)&sK[0][key][dv] = kh;
#pragma unroll
    for (int j = 0; j < 4; ++j) sVt[0][dv + j][key] = (_Float16)vv[j];
  }
  __syncthreads();

  for (int kc = 0; kc < NCHUNK; ++kc) {
    const int cur = kc & 1;
    const int nxt = cur ^ 1;
    const int kstart = kc * KC;
    const bool more = (kc + 1 < NCHUNK);

    // ---- issue next chunk's global loads (in flight during compute) ----
    v4f kreg[4], vreg[4];
    if (more) {
      const int ks = kstart + KC;
#pragma unroll
      for (int i = 0; i < 4; ++i) {
        int vec = tid + i * 256;
        int key = vec >> 5;
        int dv  = (vec & 31) << 2;
        kreg[i] = *(const v4f*)(kbase + (size_t)(ks + key) * rowstride + dv);
        vreg[i] = *(const v4f*)(vbase + (size_t)(ks + key) * rowstride + dv);
      }
    }

    // ---- S = Q K^T for two 16-key tiles ----
    v8f p[2];
    float keep[2];
#pragma unroll
    for (int t = 0; t < 2; ++t) {
      v8f s = (v8f)0.0f;
#pragma unroll
      for (int c = 0; c < 4; ++c) {
        v8h lo = *(const v8h*)&sK[cur][t * 16 + n16][c * 32 + 8 * half];
        v8h hi = *(const v8h*)&sK[cur][t * 16 + n16][c * 32 + 16 + 8 * half];
        s = __builtin_amdgcn_wmma_f32_16x16x32_f16(
            false, aq[c], false, cat8(lo, hi), (short)0, s, false, false);
      }
      int key  = kstart + t * 16 + n16;
      bool ok  = kpm[b * SS + key] != 0;
      keep[t]  = ok ? 1.0f : 0.0f;
      float bias = ok ? 0.0f : -1e30f;
#pragma unroll
      for (int r = 0; r < 8; ++r) s[r] += bias;
      p[t] = s;
    }

    // ---- retire K staging registers into the next buffer ----
    if (more) {
#pragma unroll
      for (int i = 0; i < 4; ++i) {
        int vec = tid + i * 256;
        int key = vec >> 5;
        int dv  = (vec & 31) << 2;
        v4h kh;
#pragma unroll
        for (int j = 0; j < 4; ++j) kh[j] = (_Float16)kreg[i][j];
        *(v4h*)&sK[nxt][key][dv] = kh;
      }
    }

    // ---- online softmax: row = (reg r, lane half); DPP rotate reductions --
    float cm[8];
#pragma unroll
    for (int r = 0; r < 8; ++r)
      cm[r] = rowmax16(fmaxf(p[0][r], p[1][r]));

    float corr[8];
#pragma unroll
    for (int r = 0; r < 8; ++r) {
      float nm = fmaxf(rmax[r], cm[r]);
      corr[r]  = exp2f(rmax[r] - nm);
      rmax[r]  = nm;
    }
#pragma unroll
    for (int t = 0; t < 2; ++t)
#pragma unroll
      for (int r = 0; r < 8; ++r)
        p[t][r] = exp2f(p[t][r] - rmax[r]) * keep[t];

#pragma unroll
    for (int r = 0; r < 8; ++r) {
      float cs = rowsum16(p[0][r] + p[1][r]);
      rsum[r]  = rsum[r] * corr[r] + cs;
    }
#pragma unroll
    for (int t = 0; t < 8; ++t)
#pragma unroll
      for (int r = 0; r < 8; ++r) o[t][r] *= corr[r];

    // ---- C-layout -> A-layout transpose of P via per-wave LDS tile ----
#pragma unroll
    for (int t = 0; t < 2; ++t)
#pragma unroll
      for (int r = 0; r < 8; ++r)
        sP[wave][r + 8 * half][t * 16 + n16] = (_Float16)p[t][r];
    asm volatile("s_wait_dscnt 0" ::: "memory");  // wave-local visibility

    v8h plo = *(const v8h*)&sP[wave][n16][8 * half];
    v8h phi = *(const v8h*)&sP[wave][n16][16 + 8 * half];
    v16h ap = cat8(plo, phi);

    // ---- O += P V : 8 output tiles across D=128 ----
#pragma unroll
    for (int td = 0; td < 8; ++td) {
      v8h lo = *(const v8h*)&sVt[cur][td * 16 + n16][8 * half];
      v8h hi = *(const v8h*)&sVt[cur][td * 16 + n16][16 + 8 * half];
      o[td] = __builtin_amdgcn_wmma_f32_16x16x32_f16(
          false, ap, false, cat8(lo, hi), (short)0, o[td], false, false);
    }

    // ---- retire V staging registers (transposed) into the next buffer ----
    if (more) {
#pragma unroll
      for (int i = 0; i < 4; ++i) {
        int vec = tid + i * 256;
        int key = vec >> 5;
        int dv  = (vec & 31) << 2;
#pragma unroll
        for (int j = 0; j < 4; ++j)
          sVt[nxt][dv + j][key] = (_Float16)vreg[i][j];
      }
    }
    __syncthreads();
  }

  // ---- normalize and store ----
  float* obase = out + ((size_t)(b * SS + q0) * HH + h) * DD;
#pragma unroll
  for (int r = 0; r < 8; ++r) {
    int m = r + 8 * half;
    float inv = 1.0f / rsum[r];
#pragma unroll
    for (int td = 0; td < 8; ++td)
      obase[(size_t)m * rowstride + td * 16 + n16] = o[td][r] * inv;
  }
}

extern "C" void kernel_launch(void* const* d_in, const int* in_sizes, int n_in,
                              void* d_out, int out_size, void* d_ws, size_t ws_size,
                              hipStream_t stream) {
  (void)in_sizes; (void)n_in; (void)out_size; (void)d_ws; (void)ws_size;
  const float* q = (const float*)d_in[0];
  const float* k = (const float*)d_in[1];
  const float* v = (const float*)d_in[2];
  const unsigned char* kpm = (const unsigned char*)d_in[3];  // jax bool = 1 byte
  float* out = (float*)d_out;

  dim3 grid(BB * HH * (SS / QROWS));  // 512 blocks
  dim3 block(256);                    // 8 waves, one 16-row Q tile each
  fa_fwd_f16wmma<<<grid, block, 0, stream>>>(q, k, v, kpm, out);
}
